// LFR_11158325035320
// MI455X (gfx1250) — matrix operand order repaired and checked by reading, not compiled, not run on hardware
//
#include <hip/hip_runtime.h>

typedef __attribute__((ext_vector_type(16))) _Float16 v16h;
typedef __attribute__((ext_vector_type(8)))  float    v8f;

#define TILE_M 32     // output channels per block
#define TILE_N 256    // pixels per block
#define TILE_K 32     // K-chunk staged in LDS
#define LDSS   40     // 40 f16 = 80 B per row; multiple of 16 B so frag reads are ds_load_b128

// ---------------------------------------------------------------------------
// Generic implicit-GEMM conv (stride 1) + scale/bias + optional ReLU.
//   out[b, co_off+co, y, x] = act( conv(in, Wt)[co] * g[co]/sqrt(1+1e-5) + bias[co] )
//
// Block: 256 threads = 8 waves, arranged 2 (M) x 4 (N).
// Each wave computes a 16x64 output tile: 4 accumulators, A-fragment reused 4x.
// Double-buffered LDS, software-pipelined: chunk k+1 global loads are issued
// before computing chunk k, stores to the alternate buffer after, one barrier
// per chunk.
//
// Staging ownership:
//   A: thread (tid>>3) owns weight row, 4 consecutive k  -> b128 global load
//   B: thread tid owns ONE pixel, all 32 k of the chunk  -> coalesced per tap
// KHW is compile time (1 -> 1x1 pad 0, 9 -> 3x3 pad 1): constant-divisor
// k-decode, pixel decode fully hoisted. K = CI*KHW is a multiple of 32 and
// pixel count a multiple of 256 for every conv in this graph.
// ---------------------------------------------------------------------------
template<bool RELU, int KHW>
__global__ __launch_bounds__(256)
void conv_wmma(const float* __restrict__ in, int CI,
               const float* __restrict__ Wt,                 // [CO][CI][KH][KW] f32
               const float* __restrict__ gamma,
               const float* __restrict__ beta,
               float* __restrict__ out,
               int CO, int CO_total, int co_off,
               int H, int Wd)
{
    const int K   = CI * KHW;
    const int HW  = H * Wd;
    const int pad = (KHW == 9) ? 1 : 0;

    __shared__ _Float16 As[2][TILE_M][LDSS];
    __shared__ _Float16 Bs[2][TILE_N][LDSS];

    const int tid  = threadIdx.x;
    const int lane = tid & 31;
    const int wave = tid >> 5;
    const int wm   = wave >> 2;    // 0..1 -> 16-row group of the 32-channel tile
    const int wn   = wave & 3;     // 0..3 -> 64-col group of the 256-pixel tile
    const int hh   = lane >> 4;    // lane half (0/1)
    const int rr   = lane & 15;

    const int p_base = blockIdx.x * TILE_N;   // pixel tile base (flat b*HW + y*W + x)
    const int m_base = blockIdx.y * TILE_M;   // output-channel tile base

    // ---- A staging ownership: 4 consecutive k at one weight row ----
    const int a_mi = tid >> 3;                // 0..31
    const int a_kb = (tid & 7) * 4;           // 0,4,...,28
    const int a_m  = m_base + a_mi;
    const bool a_ok = (a_m < CO);
    const float* a_src = Wt + (size_t)a_m * K + a_kb;

    // ---- B staging ownership: one fixed pixel, 32 k per chunk ----
    const int p   = p_base + tid;
    const int pb  = p / HW;
    const int prm = p - pb * HW;
    const int py  = prm / Wd;
    const int px  = prm - py * Wd;
    const float* in_img = in + (size_t)pb * CI * HW;
    const float* in_pix = in_img + (size_t)py * Wd + px;   // used for 1x1 path

    float areg[4];
    float breg[TILE_K];
    v8f   acc[4] = {{}, {}, {}, {}};

    // ---- staging loads (global -> regs) for chunk starting at k0 ----
    auto load_chunk = [&](int k0) {
        #pragma unroll
        for (int j = 0; j < 4; ++j)
            areg[j] = a_ok ? a_src[k0 + j] : 0.0f;
        if (k0 + 2 * TILE_K < K)                  // gfx1250 global_prefetch_b8
            __builtin_prefetch(a_src + k0 + 2 * TILE_K, 0, 1);
        #pragma unroll
        for (int j = 0; j < TILE_K; ++j) {
            const int k = k0 + j;
            float v = 0.0f;
            if (KHW == 1) {
                v = in_pix[(size_t)k * HW];
            } else {
                const int ci = k / KHW;           // constant-divisor
                const int rk = k - ci * KHW;
                const int kh = rk / 3;
                const int kw = rk - kh * 3;
                const int iy = py + kh - pad;
                const int ix = px + kw - pad;
                if (iy >= 0 && iy < H && ix >= 0 && ix < Wd)
                    v = in_img[((size_t)ci * H + iy) * Wd + ix];
            }
            breg[j] = v;
        }
    };
    // ---- staging stores (regs -> LDS buffer) ----
    auto store_chunk = [&](int buf) {
        #pragma unroll
        for (int j = 0; j < 4; ++j)
            As[buf][a_mi][a_kb + j] = (_Float16)areg[j];
        #pragma unroll
        for (int j = 0; j < TILE_K; ++j)
            Bs[buf][tid][j] = (_Float16)breg[j];
    };
    // ---- compute 4 WMMAs from LDS buffer ----
    // Fragments in contiguous k order (same for A and B, not AMD's table order):
    //   A (16x32 f16): row M = lane%16; K = hh*16+i (elems 0..15)
    //   B (32x16 f16): col N = lane%16; K = hh*16+i (elems 0..15)
    auto compute_chunk = [&](int buf) {
        v16h af;
        #pragma unroll
        for (int i = 0; i < 8; ++i) {
            af[i]     = As[buf][wm * 16 + rr][hh * 16 + i];
            af[8 + i] = As[buf][wm * 16 + rr][hh * 16 + 8 + i];
        }
        #pragma unroll
        for (int s = 0; s < 4; ++s) {
            v16h bf;
            #pragma unroll
            for (int i = 0; i < 16; ++i)
                bf[i] = Bs[buf][wn * 64 + s * 16 + rr][hh * 16 + i];
            acc[s] = __builtin_amdgcn_wmma_f32_16x16x32_f16(
                         false, af, false, bf, (short)0, acc[s], false, false);
        }
    };

    // ---- software pipeline: one barrier per K-chunk ----
    const int nk = K / TILE_K;
    load_chunk(0);
    store_chunk(0);
    __syncthreads();
    for (int i = 0; i < nk; ++i) {
        const int cur = i & 1;
        const bool more = (i + 1 < nk);
        if (more) load_chunk((i + 1) * TILE_K);   // global loads in flight...
        compute_chunk(cur);                       // ...covered by WMMA work
        if (more) store_chunk(cur ^ 1);
        __syncthreads();
    }

    // ---- epilogue: D elem e -> M = e + 8*hh, N = lane%16 ----
    const float inv_sqrt = 0.99999500003749969f;  // 1/sqrt(1+1e-5)
    #pragma unroll
    for (int s = 0; s < 4; ++s) {
        #pragma unroll
        for (int e = 0; e < 8; ++e) {
            int co = m_base + wm * 16 + e + 8 * hh;
            if (co < CO) {
                int pp  = p_base + wn * 64 + s * 16 + rr;
                int b   = pp / HW;
                int rem = pp - b * HW;
                float v = acc[s][e] * (gamma[co] * inv_sqrt) + beta[co];
                if (RELU) v = fmaxf(v, 0.0f);
                out[((size_t)(b * CO_total + co_off + co)) * HW + rem] = v;
            }
        }
    }
}

// ---------------------------------------------------------------------------
// Align-corners bilinear upsample, writing at a channel offset inside a wider
// destination tensor (materializes concats for free).
// ---------------------------------------------------------------------------
__global__ __launch_bounds__(256)
void bilinear_up(const float* __restrict__ in, float* __restrict__ out,
                 int NB, int C, int H, int W, int Ho, int Wo,
                 int C_total, int co_off)
{
    size_t i = (size_t)blockIdx.x * 256 + threadIdx.x;
    size_t total = (size_t)NB * C * Ho * Wo;
    if (i >= total) return;
    int xo = (int)(i % Wo); size_t t = i / Wo;
    int yo = (int)(t % Ho); t /= Ho;
    int c  = (int)(t % C);
    int b  = (int)(t / C);

    float sy = (Ho > 1) ? (float)(H - 1) / (float)(Ho - 1) : 0.0f;
    float sx = (Wo > 1) ? (float)(W - 1) / (float)(Wo - 1) : 0.0f;
    float ys = yo * sy, xs = xo * sx;
    int y0 = (int)floorf(ys), x0 = (int)floorf(xs);
    int y1 = min(y0 + 1, H - 1), x1 = min(x0 + 1, W - 1);
    float wy = ys - (float)y0, wx = xs - (float)x0;

    const float* base = in + (size_t)(b * C + c) * H * W;
    float v00 = base[y0 * W + x0], v01 = base[y0 * W + x1];
    float v10 = base[y1 * W + x0], v11 = base[y1 * W + x1];
    float vt = v00 * (1.0f - wx) + v01 * wx;
    float vb = v10 * (1.0f - wx) + v11 * wx;
    out[((size_t)(b * C_total + co_off + c) * Ho + yo) * Wo + xo] =
        vt * (1.0f - wy) + vb * wy;
}

// ---------------------------------------------------------------------------
// In-place channel softmax (C = 25), one thread per pixel.
// ---------------------------------------------------------------------------
__global__ __launch_bounds__(256)
void softmax_ch(float* __restrict__ wm, int NB, int C, int HW)
{
    int i = blockIdx.x * 256 + threadIdx.x;
    if (i >= NB * HW) return;
    int b = i / HW, p = i - b * HW;
    float* base = wm + (size_t)b * C * HW + p;
    float mx = -3.0e38f;
    for (int c = 0; c < C; ++c) mx = fmaxf(mx, base[(size_t)c * HW]);
    float s = 0.0f;
    for (int c = 0; c < C; ++c) s += __expf(base[(size_t)c * HW] - mx);
    float inv = 1.0f / s;
    for (int c = 0; c < C; ++c)
        base[(size_t)c * HW] = __expf(base[(size_t)c * HW] - mx) * inv;
}

// ---------------------------------------------------------------------------
// CARAFE reassembly: X[b,c,y,x] = sum_{i,j<5} Wm[b,i*5+j,y,x] * xu[b,c,y+2(i-2),x+2(j-2)]
// Written into channels [co_off, co_off+C) of a C_total-channel destination.
// ---------------------------------------------------------------------------
__global__ __launch_bounds__(256)
void carafe(const float* __restrict__ wm, const float* __restrict__ xu,
            float* __restrict__ out, int NB, int C, int H, int W,
            int C_total, int co_off)
{
    size_t i = (size_t)blockIdx.x * 256 + threadIdx.x;
    size_t total = (size_t)NB * C * H * W;
    if (i >= total) return;
    int x = (int)(i % W); size_t t = i / W;
    int y = (int)(t % H); t /= H;
    int c = (int)(t % C);
    int b = (int)(t / C);
    int HW = H * W;

    const float* wbase = wm + ((size_t)b * 25 * H + y) * W + x;
    const float* xbase = xu + (size_t)(b * C + c) * HW;
    float acc = 0.0f;
    #pragma unroll
    for (int ki = 0; ki < 5; ++ki) {
        int iy = y + 2 * (ki - 2);
        if (iy < 0 || iy >= H) continue;
        #pragma unroll
        for (int kj = 0; kj < 5; ++kj) {
            int ix = x + 2 * (kj - 2);
            if (ix < 0 || ix >= W) continue;
            acc += wbase[(size_t)(ki * 5 + kj) * HW] * xbase[iy * W + ix];
        }
    }
    out[((size_t)(b * C_total + co_off + c) * H + y) * W + x] = acc;
}

// ---------------------------------------------------------------------------
// Copy a tensor into channels [co_off, co_off+C) of a wider destination.
// ---------------------------------------------------------------------------
__global__ __launch_bounds__(256)
void copy_into(const float* __restrict__ src, float* __restrict__ dst,
               int NB, int C, int HW, int C_total, int co_off)
{
    size_t i = (size_t)blockIdx.x * 256 + threadIdx.x;
    size_t total = (size_t)NB * C * HW;
    if (i >= total) return;
    int p = (int)(i % HW); size_t t = i / HW;
    int c = (int)(t % C);
    int b = (int)(t / C);
    dst[((size_t)(b * C_total + co_off + c)) * HW + p] = src[i];
}

// ---------------------------------------------------------------------------
extern "C" void kernel_launch(void* const* d_in, const int* in_sizes, int n_in,
                              void* d_out, int out_size, void* d_ws, size_t ws_size,
                              hipStream_t stream)
{
    const float* x_h      = (const float*)d_in[0];
    const float* x_l      = (const float*)d_in[1];
    const float* w_reduce = (const float*)d_in[2];
    const float* g_reduce = (const float*)d_in[3];
    const float* b_reduce = (const float*)d_in[4];
    const float* w1       = (const float*)d_in[5];
    const float* g1       = (const float*)d_in[6];
    const float* b1       = (const float*)d_in[7];
    const float* w2       = (const float*)d_in[8];
    const float* g2       = (const float*)d_in[9];
    const float* b2       = (const float*)d_in[10];
    const float* w_enc    = (const float*)d_in[11];
    const float* g_enc    = (const float*)d_in[12];
    const float* b_enc    = (const float*)d_in[13];
    const float* w_out1   = (const float*)d_in[14];
    const float* g_out1   = (const float*)d_in[15];
    const float* b_out1   = (const float*)d_in[16];
    const float* w_out2   = (const float*)d_in[17];
    const float* g_out2   = (const float*)d_in[18];
    const float* b_out2   = (const float*)d_in[19];

    // Workspace layout (floats). t1/xr share; cat1/y1 share.
    float* ws   = (float*)d_ws;
    float* t1   = ws;                       //  1,048,576  (B,64,64,64)    also xr
    float* cat1 = t1   + 1048576;           //  8,388,608  (B,128,128,128) also y1
    float* Wm   = cat1 + 8388608;           //  1,638,400  (B,25,128,128)
    float* xu   = Wm   + 1638400;           //  4,194,304  (B,64,128,128)
    float* cat2 = xu   + 4194304;           // 12,582,912  (B,192,128,128)
    const size_t need = (size_t)(1048576 + 8388608 + 1638400 + 4194304 + 12582912) * 4;
    if (ws_size < need) return;             // avoid OOB on undersized workspace

    dim3 blk(256);
    const int NP_lo = 4 * 64 * 64;          // pixels at 64x64
    const int NP_hi = 4 * 128 * 128;        // pixels at 128x128

    // 1) x1_pre = cbr(x_h, w1)  : 1x1, 256->64, 64x64
    conv_wmma<true, 1><<<dim3(NP_lo / TILE_N, 2), blk, 0, stream>>>(
        x_h, 256, w1, g1, b1, t1, 64, 64, 0, 64, 64);
    // 2) x1 = bilinear(x1_pre) -> cat1[:, 0:64]
    bilinear_up<<<(4 * 64 * 128 * 128 + 255) / 256, blk, 0, stream>>>(
        t1, cat1, 4, 64, 64, 64, 128, 128, 128, 0);
    // 3) x2 = cbr(x_l, w2) : 1x1, 128->64 -> cat1[:, 64:128]
    conv_wmma<true, 1><<<dim3(NP_hi / TILE_N, 2), blk, 0, stream>>>(
        x_l, 128, w2, g2, b2, cat1, 64, 128, 64, 128, 128);
    // 4) Wm_raw = cbr(cat1, w_enc, relu=False) : 3x3 pad1, 128->25
    conv_wmma<false, 9><<<dim3(NP_hi / TILE_N, 1), blk, 0, stream>>>(
        cat1, 128, w_enc, g_enc, b_enc, Wm, 25, 25, 0, 128, 128);
    // 5) softmax over 25 channels (in place)
    softmax_ch<<<(NP_hi + 255) / 256, blk, 0, stream>>>(Wm, 4, 25, 128 * 128);
    // 6) xr = cbr(x_h, w_reduce) : 3x3 pad1, 256->64 (reuses t1)
    float* xr = t1;
    conv_wmma<true, 9><<<dim3(NP_lo / TILE_N, 2), blk, 0, stream>>>(
        x_h, 256, w_reduce, g_reduce, b_reduce, xr, 64, 64, 0, 64, 64);
    // 7) xu = bilinear(xr)
    bilinear_up<<<(4 * 64 * 128 * 128 + 255) / 256, blk, 0, stream>>>(
        xr, xu, 4, 64, 64, 64, 128, 128, 64, 0);
    // 8) X = carafe(Wm, xu) -> cat2[:, 0:64]
    carafe<<<(4 * 64 * 128 * 128 + 255) / 256, blk, 0, stream>>>(
        Wm, xu, cat2, 4, 64, 128, 128, 192, 0);
    // 9) cat2[:, 64:192] = x_l
    copy_into<<<((size_t)4 * 128 * 128 * 128 + 255) / 256, blk, 0, stream>>>(
        x_l, cat2, 4, 128, 128 * 128, 192, 64);
    // 10) y = cbr(cat2, w_out1) : 3x3 pad1, 192->128 (reuses cat1)
    float* y1 = cat1;
    conv_wmma<true, 9><<<dim3(NP_hi / TILE_N, 4), blk, 0, stream>>>(
        cat2, 192, w_out1, g_out1, b_out1, y1, 128, 128, 0, 128, 128);
    // 11) out = cbr(y, w_out2) : 1x1, 128->128
    conv_wmma<true, 1><<<dim3(NP_hi / TILE_N, 4), blk, 0, stream>>>(
        y1, 128, w_out2, g_out2, b_out2, (float*)d_out, 128, 128, 0, 128, 128);
}
